// HackableDecoderLayer_7705171329764
// MI455X (gfx1250) — compile-verified, matrix operand order checked by reading, not run on hardware
//
#include <hip/hip_runtime.h>
#include <hip/hip_bf16.h>
#include <math.h>

// ---------------------------------------------------------------------------
// Problem constants (reference: B=2,S=2048,E=1024,KVH=4,QPG=4,H=64,MLP=4096)
// ---------------------------------------------------------------------------
#define BB   2
#define SS   2048
#define EE   1024
#define KVHN 4
#define QPGN 4
#define HH   64
#define MLPN 4096
#define NQH  16      // KVH*QPG

typedef __attribute__((ext_vector_type(16))) __bf16 v16bf;
typedef __attribute__((ext_vector_type(8)))  __bf16 v8bf;
typedef __attribute__((ext_vector_type(8)))  float  v8f;

__device__ __forceinline__ __bf16 f2bf(float f) {
  unsigned u = __builtin_bit_cast(unsigned, f);
  unsigned r = (u + 0x7FFFu + ((u >> 16) & 1u)) >> 16;   // round-to-nearest-even
  unsigned short s = (unsigned short)r;
  return __builtin_bit_cast(__bf16, s);
}

__device__ __forceinline__ v8f vzero8() {
  v8f z;
#pragma unroll
  for (int i = 0; i < 8; ++i) z[i] = 0.f;
  return z;
}

__device__ __forceinline__ v16bf cat8(v8bf lo, v8bf hi) {
  return __builtin_shufflevector(lo, hi, 0, 1, 2, 3, 4, 5, 6, 7,
                                 8, 9, 10, 11, 12, 13, 14, 15);
}

__device__ __forceinline__ v8f wmma_bf16(v16bf a, v16bf b, v8f c) {
  // D = A(16x32 bf16) * B(32x16 bf16) + C(16x16 f32)
  return __builtin_amdgcn_wmma_f32_16x16x32_bf16(false, a, false, b,
                                                 (short)0, c, false, false);
}

// 16x32 bf16 fragment from a bf16 row: chunks at hi*8 and 16+hi*8 (ISA layout)
__device__ __forceinline__ v16bf load_frag_row_bf16(const __bf16* __restrict__ row,
                                                    int kk, int hi) {
  const v8bf* a = (const v8bf*)(row + kk * 32 + hi * 8);
  const v8bf* b = (const v8bf*)(row + kk * 32 + 16 + hi * 8);
  return cat8(*a, *b);
}

// ---------------------------------------------------------------------------
// Async LDS-DMA: copy 64B/lane global -> LDS (4x B128, ASYNCcnt-tracked).
// IOFFSET is added to BOTH the global and the LDS address (ISA 10.x async),
// so one base pair covers the 64-byte chunk.
// ---------------------------------------------------------------------------
__device__ __forceinline__ void async_copy64(void* lds, const __bf16* g) {
  unsigned           l  = (unsigned)(uintptr_t)lds;       // low 32b = LDS offset
  unsigned long long ga = (unsigned long long)(uintptr_t)g;
  asm volatile(
      "global_load_async_to_lds_b128 %0, %1, off\n\t"
      "global_load_async_to_lds_b128 %0, %1, off offset:16\n\t"
      "global_load_async_to_lds_b128 %0, %1, off offset:32\n\t"
      "global_load_async_to_lds_b128 %0, %1, off offset:48"
      :: "v"(l), "v"(ga) : "memory");
}

// ---------------------------------------------------------------------------
// One-time fp32 -> bf16 weight conversion (streaming, 8 elems/thread)
// ---------------------------------------------------------------------------
__global__ __launch_bounds__(256)
void f32_to_bf16(const float* __restrict__ in, __bf16* __restrict__ out) {
  const size_t i = ((size_t)blockIdx.x * 256 + threadIdx.x) * 8;
  const float4* s = (const float4*)(in + i);
  float4 f0 = s[0], f1 = s[1];
  v8bf p;
  p[0] = f2bf(f0.x); p[1] = f2bf(f0.y); p[2] = f2bf(f0.z); p[3] = f2bf(f0.w);
  p[4] = f2bf(f1.x); p[5] = f2bf(f1.y); p[6] = f2bf(f1.z); p[7] = f2bf(f1.w);
  *(v8bf*)(out + i) = p;
}

// ---------------------------------------------------------------------------
// RMSNorm (fp32 in) -> bf16 out, one block per row of E=1024
// ---------------------------------------------------------------------------
__global__ __launch_bounds__(256)
void rmsnorm_bf16(const float* __restrict__ x, const float* __restrict__ w,
                  __bf16* __restrict__ out) {
  const int row = blockIdx.x;
  const int tid = threadIdx.x;
  const float* xr = x + (size_t)row * EE;
  __shared__ float red[8];
  float ss = 0.f;
#pragma unroll
  for (int i = 0; i < EE / 256; ++i) {
    float v = xr[tid + i * 256];
    ss += v * v;
  }
#pragma unroll
  for (int off = 16; off > 0; off >>= 1) ss += __shfl_xor(ss, off, 32);
  if ((tid & 31) == 0) red[tid >> 5] = ss;
  __syncthreads();
  float tot = 0.f;
#pragma unroll
  for (int i = 0; i < 8; ++i) tot += red[i];
  const float rms = rsqrtf(tot * (1.0f / EE) + 1e-5f);
#pragma unroll
  for (int i = 0; i < EE / 256; ++i) {
    int e = tid + i * 256;
    out[(size_t)row * EE + e] = f2bf(xr[e] * rms * w[e]);
  }
}

// ---------------------------------------------------------------------------
// Tiled WMMA GEMM:  C[M,N] = A[M,K] * W[N,K]^T  (+ fused epilogues)
// A and W both bf16.  Block = 256 thr (8 waves, 4x2), tile 128x128.
// K staged 64 at a time via ASYNC LDS-DMA, 2-deep ping-pong pipeline:
//   issue stage k+1, s_wait_asynccnt 8 (stage k done, in-order), barrier,
//   16 WMMAs from LDS, barrier.
// ---------------------------------------------------------------------------
enum : int { EPI_F32 = 0, EPI_RESID = 1, EPI_RELU2BF16 = 2, EPI_QKROPE = 3,
             EPI_VT = 4 };

template <int EPI>
__global__ __launch_bounds__(256, 2)
void gemm_wmma(const __bf16* __restrict__ A, const __bf16* __restrict__ W,
               const float* __restrict__ resid, const float* __restrict__ nw,
               float oscale, void* __restrict__ outv, int M, int N, int K) {
  constexpr int LDT = 72;                       // padded row stride (bf16 elems)
  __shared__ __bf16 As[2][128 * LDT];
  __shared__ __bf16 Bs[2][128 * LDT];

  const int tid  = threadIdx.x;
  const int lane = tid & 31;
  const int wave = tid >> 5;
  const int wrow = wave >> 1;                   // 0..3 (M)
  const int wcol = wave & 1;                    // 0..1 (N)
  const int lm   = lane & 15;
  const int hi   = lane >> 4;

  const int m0 = blockIdx.y * 128;
  const int n0 = blockIdx.x * 128;

  v8f acc[2][4];
#pragma unroll
  for (int i = 0; i < 2; ++i)
#pragma unroll
    for (int j = 0; j < 4; ++j) acc[i][j] = vzero8();

  const int lr = tid >> 1;                      // row 0..127 of the tile
  const int lc = (tid & 1) * 32;                // col 0 or 32

  const __bf16* gA = A + (size_t)(m0 + lr) * K + lc;
  const __bf16* gW = W + (size_t)(n0 + lr) * K + lc;
  const int ldsoff = lr * LDT + lc;

  // prologue: stage 0 into buffer 0
  async_copy64(&As[0][ldsoff], gA);
  async_copy64(&Bs[0][ldsoff], gW);

  const int nstage = K >> 6;
  for (int st = 0; st < nstage; ++st) {
    const int buf = st & 1;
    if (st + 1 < nstage) {
      // prefetch next stage into the other buffer, then wait for this stage
      async_copy64(&As[buf ^ 1][ldsoff], gA + (st + 1) * 64);
      async_copy64(&Bs[buf ^ 1][ldsoff], gW + (st + 1) * 64);
      asm volatile("s_wait_asynccnt 0x8" ::: "memory");
    } else {
      asm volatile("s_wait_asynccnt 0x0" ::: "memory");
    }
    __syncthreads();

#pragma unroll
    for (int ks = 0; ks < 2; ++ks) {
      v16bf af[2], bfr[4];
#pragma unroll
      for (int mi = 0; mi < 2; ++mi) {
        const __bf16* p =
            &As[buf][(wrow * 32 + mi * 16 + lm) * LDT + ks * 32 + hi * 8];
        af[mi] = cat8(*(const v8bf*)p, *(const v8bf*)(p + 16));
      }
#pragma unroll
      for (int ni = 0; ni < 4; ++ni) {
        const __bf16* p =
            &Bs[buf][(wcol * 64 + ni * 16 + lm) * LDT + ks * 32 + hi * 8];
        bfr[ni] = cat8(*(const v8bf*)p, *(const v8bf*)(p + 16));
      }
#pragma unroll
      for (int mi = 0; mi < 2; ++mi)
#pragma unroll
        for (int ni = 0; ni < 4; ++ni)
          acc[mi][ni] = wmma_bf16(af[mi], bfr[ni], acc[mi][ni]);
    }
    __syncthreads();
  }

  // ---- epilogues (C layout: VGPR r -> M = hi*8+r, N = lm) ----
  if (EPI == EPI_QKROPE) {
    const int nb = n0 + wcol * 64;              // head base (multiple of 64)
    // theta=1024 -> inv_freq(lm) = 2^(-10*lm/16); hardware transcendentals
    const float fw = __builtin_amdgcn_exp2f((float)lm * (-10.0f / 16.0f));
    const float w0 = nw[lm], w1 = nw[16 + lm], w2 = nw[32 + lm], w3 = nw[48 + lm];
#pragma unroll
    for (int mi = 0; mi < 2; ++mi) {
#pragma unroll
      for (int r = 0; r < 8; ++r) {
        const int m = m0 + wrow * 32 + mi * 16 + hi * 8 + r;
        const float v0 = acc[mi][0][r], v1 = acc[mi][1][r];
        const float v2 = acc[mi][2][r], v3 = acc[mi][3][r];
        float ssq = v0 * v0 + v1 * v1 + v2 * v2 + v3 * v3;
#pragma unroll
        for (int off = 8; off > 0; off >>= 1) ssq += __shfl_xor(ssq, off, 32);
        const float rms = rsqrtf(ssq * (1.0f / HH) + 1e-5f);
        const float t0 = v0 * rms * w0, t1 = v1 * rms * w1;
        const float t2 = v2 * rms * w2, t3 = v3 * rms * w3;
        const int pos = m & (SS - 1);
        const float ang = (float)pos * fw;
        const float c = __cosf(ang), s = __sinf(ang);   // v_cos_f32 / v_sin_f32
        __bf16* ob = (__bf16*)outv;
        const size_t base = (size_t)m * N + nb + lm;
        ob[base]      = f2bf((t0 * c - t2 * s) * oscale);  // d in [0,16)
        ob[base + 16] = f2bf(t1 * oscale);                 // d in [16,32): id
        ob[base + 32] = f2bf((t2 * c + t0 * s) * oscale);  // d in [32,48)
        ob[base + 48] = f2bf(t3 * oscale);                 // d in [48,64): id
      }
    }
  } else {
#pragma unroll
    for (int mi = 0; mi < 2; ++mi) {
#pragma unroll
      for (int ni = 0; ni < 4; ++ni) {
#pragma unroll
        for (int r = 0; r < 8; ++r) {
          const int m = m0 + wrow * 32 + mi * 16 + hi * 8 + r;
          const int n = n0 + wcol * 64 + ni * 16 + lm;
          const float v = acc[mi][ni][r];
          if (EPI == EPI_F32) {
            ((float*)outv)[(size_t)m * N + n] = v;
          } else if (EPI == EPI_RESID) {
            const size_t idx = (size_t)m * N + n;
            ((float*)outv)[idx] = v + resid[idx];
          } else if (EPI == EPI_RELU2BF16) {
            const float rr = v > 0.f ? v : 0.f;
            ((__bf16*)outv)[(size_t)m * N + n] = f2bf(rr * rr);
          } else {  // EPI_VT: out[b][n][s] = acc, b = m/S, s = m%S
            const int bb = m >> 11, s = m & (SS - 1);
            ((__bf16*)outv)[((size_t)bb * N + n) * SS + s] = f2bf(v);
          }
        }
      }
    }
  }
}

// ---------------------------------------------------------------------------
// Flash attention: grid (B*16 heads, S/128), 8 waves x 16 Q-rows per block.
// Q/K/V all bf16 (scale folded into Q at QKROPE epilogue).
// Key blocks of 32: QK^T = 4 WMMAs, online softmax, P*V = 4 WMMAs.
// ---------------------------------------------------------------------------
__global__ __launch_bounds__(256, 2)
void flash_attn(const __bf16* __restrict__ q, const __bf16* __restrict__ k,
                const __bf16* __restrict__ vt, __bf16* __restrict__ attn) {
  __shared__ __bf16 Pl[8][16 * 40];        // per-wave 16x32 P tile (padded)

  const int tid  = threadIdx.x;
  const int lane = tid & 31;
  const int wave = tid >> 5;
  const int lm   = lane & 15;
  const int hi   = lane >> 4;

  const int hb   = blockIdx.x;             // b*16 + head
  const int b    = hb >> 4;
  const int head = hb & 15;
  const int kh   = head >> 2;              // head = kh*QPG + g

  const int q0 = blockIdx.y * 128 + wave * 16;

  // Q fragments for this wave's 16 rows (two K=32 steps over H=64)
  v16bf qf[2];
  {
    const __bf16* qrow = q + ((size_t)(b * SS + q0 + lm) * NQH + head) * HH;
#pragma unroll
    for (int kk = 0; kk < 2; ++kk) qf[kk] = load_frag_row_bf16(qrow, kk, hi);
  }

  float mrow[8], lrow[8];
#pragma unroll
  for (int r = 0; r < 8; ++r) { mrow[r] = -INFINITY; lrow[r] = 0.f; }
  v8f o[4];
#pragma unroll
  for (int nn = 0; nn < 4; ++nn) o[nn] = vzero8();

  const int jbmax = (q0 + 15) >> 5;
  for (int jb = 0; jb <= jbmax; ++jb) {
    const int j0 = jb * 32;

    // ---- scores: two 16x16 tiles over 32 keys ----
    v8f s0 = vzero8(), s1 = vzero8();
    const __bf16* krow0 =
        k + ((size_t)(b * SS + j0 + lm) * KVHN + kh) * HH;
    const __bf16* krow1 =
        k + ((size_t)(b * SS + j0 + 16 + lm) * KVHN + kh) * HH;
#pragma unroll
    for (int kk = 0; kk < 2; ++kk) {
      s0 = wmma_bf16(qf[kk], load_frag_row_bf16(krow0, kk, hi), s0);
      s1 = wmma_bf16(qf[kk], load_frag_row_bf16(krow1, kk, hi), s1);
    }

    // ---- causal mask ----
#pragma unroll
    for (int r = 0; r < 8; ++r) {
      const int srow = q0 + hi * 8 + r;
      if (j0 + lm > srow)      s0[r] = -INFINITY;
      if (j0 + 16 + lm > srow) s1[r] = -INFINITY;
    }

    // ---- online softmax stats (row spread over 16 lanes of a half-wave) ----
    float mnew[8], alpha[8];
#pragma unroll
    for (int r = 0; r < 8; ++r) {
      float v = fmaxf(s0[r], s1[r]);
#pragma unroll
      for (int off = 8; off > 0; off >>= 1) v = fmaxf(v, __shfl_xor(v, off, 32));
      mnew[r]  = fmaxf(mrow[r], v);
      alpha[r] = __expf(mrow[r] - mnew[r]);
      mrow[r]  = mnew[r];
    }
    float p0[8], p1[8];
#pragma unroll
    for (int r = 0; r < 8; ++r) {
      p0[r] = __expf(s0[r] - mnew[r]);
      p1[r] = __expf(s1[r] - mnew[r]);
      float rs = p0[r] + p1[r];
#pragma unroll
      for (int off = 8; off > 0; off >>= 1) rs += __shfl_xor(rs, off, 32);
      lrow[r] = lrow[r] * alpha[r] + rs;
    }
#pragma unroll
    for (int nn = 0; nn < 4; ++nn)
#pragma unroll
      for (int r = 0; r < 8; ++r) o[nn][r] *= alpha[r];

    // ---- P: C layout -> A layout via per-wave LDS tile ----
    __bf16* pl = &Pl[wave][0];
#pragma unroll
    for (int r = 0; r < 8; ++r) {
      pl[(hi * 8 + r) * 40 + lm]      = f2bf(p0[r]);
      pl[(hi * 8 + r) * 40 + 16 + lm] = f2bf(p1[r]);
    }
    __builtin_amdgcn_wave_barrier();
    asm volatile("s_wait_dscnt 0x0" ::: "memory");
    v16bf pf;
    {
      const __bf16* pp = &pl[lm * 40 + hi * 8];
      pf = cat8(*(const v8bf*)pp, *(const v8bf*)(pp + 16));
    }
    __builtin_amdgcn_wave_barrier();

    // ---- P * V : 4 output tiles of 16 dims each ----
#pragma unroll
    for (int nn = 0; nn < 4; ++nn) {
      const __bf16* vrow =
          vt + ((size_t)(b * KVHN + kh) * HH + nn * 16 + lm) * SS + j0;
      v16bf vf = cat8(*(const v8bf*)(vrow + hi * 8),
                      *(const v8bf*)(vrow + 16 + hi * 8));
      o[nn] = wmma_bf16(pf, vf, o[nn]);
    }
  }

  // ---- normalize, store attn[b, s, head, h] as bf16 (feeds o-proj GEMM) ----
#pragma unroll
  for (int nn = 0; nn < 4; ++nn) {
#pragma unroll
    for (int r = 0; r < 8; ++r) {
      const int srow = q0 + hi * 8 + r;
      const size_t idx =
          ((size_t)(b * SS + srow) * NQH + head) * HH + nn * 16 + lm;
      attn[idx] = f2bf(o[nn][r] * __builtin_amdgcn_rcpf(lrow[r]));
    }
  }
}

// ---------------------------------------------------------------------------
// Orchestration.  Workspace (~61 MB of d_ws), all bf16 activations:
//   [ 0, 8M)  h / h2 (T x E)
//   [ 8,16M)  q (T x 1024)            -- later aliased by act
//   [16,18M)  k (T x 256)
//   [18,20M)  vt (B*KVH*H x S)
//   [20,28M)  attn (T x 1024)
//   [ 8,40M)  act (T x 4096)          -- aliases q/k/vt/attn (all dead)
//   [40,61M)  bf16 weights: wq wk wv wo w_up w_down
// Residual x1 lives in d_out (down-proj epilogue reads+writes elementwise).
// ---------------------------------------------------------------------------
extern "C" void kernel_launch(void* const* d_in, const int* in_sizes, int n_in,
                              void* d_out, int out_size, void* d_ws, size_t ws_size,
                              hipStream_t stream) {
  (void)in_sizes; (void)n_in; (void)out_size; (void)ws_size;
  const float* x   = (const float*)d_in[0];
  const float* wq  = (const float*)d_in[1];
  const float* wk  = (const float*)d_in[2];
  const float* wv  = (const float*)d_in[3];
  const float* wo  = (const float*)d_in[4];
  const float* wup = (const float*)d_in[5];
  const float* wdn = (const float*)d_in[6];
  const float* qnw = (const float*)d_in[7];
  const float* knw = (const float*)d_in[8];
  const float* ln1 = (const float*)d_in[9];
  const float* ln2 = (const float*)d_in[10];
  float* out = (float*)d_out;

  const size_t MB = (size_t)1 << 20;
  char* ws = (char*)d_ws;
  __bf16* h     = (__bf16*)(ws);
  __bf16* qbf   = (__bf16*)(ws + 8 * MB);
  __bf16* kbf   = (__bf16*)(ws + 16 * MB);
  __bf16* vt    = (__bf16*)(ws + 18 * MB);
  __bf16* attnb = (__bf16*)(ws + 20 * MB);
  __bf16* act   = (__bf16*)(ws + 8 * MB);       // aliases dead q/k/vt/attn
  __bf16* wqb   = (__bf16*)(ws + 40 * MB);
  __bf16* wkb   = (__bf16*)(ws + 42 * MB);
  __bf16* wvb   = (__bf16*)(ws + 42 * MB + 512 * 1024);
  __bf16* wob   = (__bf16*)(ws + 43 * MB);
  __bf16* wupb  = (__bf16*)(ws + 45 * MB);
  __bf16* wdnb  = (__bf16*)(ws + 53 * MB);

  const int T = BB * SS;   // 4096 token rows
  dim3 blk(256);

  // one-time weight conversion to bf16 (streaming)
  f32_to_bf16<<<(NQH * HH * EE) / 2048, blk, 0, stream>>>(wq, wqb);
  f32_to_bf16<<<(KVHN * HH * EE) / 2048, blk, 0, stream>>>(wk, wkb);
  f32_to_bf16<<<(KVHN * HH * EE) / 2048, blk, 0, stream>>>(wv, wvb);
  f32_to_bf16<<<(EE * NQH * HH) / 2048, blk, 0, stream>>>(wo, wob);
  f32_to_bf16<<<(MLPN * EE) / 2048, blk, 0, stream>>>(wup, wupb);
  f32_to_bf16<<<(EE * MLPN) / 2048, blk, 0, stream>>>(wdn, wdnb);

  // h = rms(x, ln1)
  rmsnorm_bf16<<<T, blk, 0, stream>>>(x, ln1, h);

  // q/k/v projections with fused qk-norm + RoPE / V-transpose epilogues
  gemm_wmma<EPI_QKROPE><<<dim3((NQH * HH) / 128, T / 128), blk, 0, stream>>>(
      h, wqb, nullptr, qnw, 0.125f, qbf, T, NQH * HH, EE);
  gemm_wmma<EPI_QKROPE><<<dim3((KVHN * HH) / 128, T / 128), blk, 0, stream>>>(
      h, wkb, nullptr, knw, 1.0f, kbf, T, KVHN * HH, EE);
  gemm_wmma<EPI_VT><<<dim3((KVHN * HH) / 128, T / 128), blk, 0, stream>>>(
      h, wvb, nullptr, nullptr, 1.0f, vt, T, KVHN * HH, EE);

  // causal GQA flash attention (bf16 in, bf16 out)
  flash_attn<<<dim3(BB * NQH, SS / 128), blk, 0, stream>>>(qbf, kbf, vt, attnb);

  // x1 = x + attn @ wo^T   (fp32, written to d_out)
  gemm_wmma<EPI_RESID><<<dim3(EE / 128, T / 128), blk, 0, stream>>>(
      attnb, wob, x, nullptr, 1.0f, out, T, EE, NQH * HH);

  // h2 = rms(x1, ln2)
  rmsnorm_bf16<<<T, blk, 0, stream>>>(out, ln2, h);

  // act = relu(h2 @ w_up^T)^2   (bf16)
  gemm_wmma<EPI_RELU2BF16><<<dim3(MLPN / 128, T / 128), blk, 0, stream>>>(
      h, wupb, nullptr, nullptr, 1.0f, act, T, MLPN, EE);

  // out = x1 + act @ w_down^T
  gemm_wmma<EPI_RESID><<<dim3(EE / 128, T / 128), blk, 0, stream>>>(
      act, wdnb, out, nullptr, 1.0f, out, T, EE, MLPN);
}
